// RQSpline_36618891165833
// MI455X (gfx1250) — compile-verified, matrix operand order checked by reading, not run on hardware
//
#include <hip/hip_runtime.h>

typedef __attribute__((ext_vector_type(16))) _Float16 v16h;
typedef __attribute__((ext_vector_type(8)))  _Float16 v8h;
typedef __attribute__((ext_vector_type(8)))  float    v8f;

#define B_TOT 65536
#define N_DIM 32
#define L_LAY 8
#define H_DIM 256
#define K_BIN 8
#define P_PAR 25
#define NP_   800
#define TB    64

// ---- LDS layout (bytes) ----
#define XB_OFF 0                      // f32 [64][32]   = 8192
#define MX_OFF 8192                   // f16 [64][32]   = 4096
#define HB_OFF 12288                  // f16 [64][256]  = 32768
#define RB_OFF 45056                  // f16 [64][256]  = 32768
#define PB_OFF 77824                  // f32 [64][800]  = 204800
#define AC_OFF 282624                 // f32 [64]       = 256
#define SMEM_BYTES 282880

// ---------------------------------------------------------------------------
// Weight converter: f32 row-major (Hin x Hout) -> f16 WMMA B-fragment layout.
// Fragment (l, ct, kt) = 512 contiguous halves; lane L holds 16 halves
// matching the 16-bit B operand layout (lanes 0-15: N=lane, K = k0..k0+7 and
// k0+16..k0+23; lanes 16-31: N=lane-16, K shifted by +8).
// ---------------------------------------------------------------------------
__global__ void wconv(const float* __restrict__ W, _Float16* __restrict__ out,
                      int Hin, int Hout) {
  long total = (long)L_LAY * Hin * Hout;
  int nkt = Hin >> 5, nct = Hout >> 4;
  for (long t = (long)blockIdx.x * blockDim.x + threadIdx.x; t < total;
       t += (long)gridDim.x * blockDim.x) {
    int i = (int)(t & 15);
    long r = t >> 4;
    int lane = (int)(r & 31); r >>= 5;
    int kt = (int)(r % nkt); r /= nkt;
    int ct = (int)(r % nct);
    int l  = (int)(r / nct);
    int n  = ct * 16 + (lane & 15);
    int hi = lane >> 4;
    int k  = kt * 32 + (i < 8 ? i : i + 8) + hi * 8;
    out[t] = (_Float16)W[((long)l * Hin + k) * Hout + n];
  }
}

// ---- A fragment from LDS f16 row-major buffer (16-bit A layout, wave32) ----
__device__ __forceinline__ v16h load_a_frag(const _Float16* base, int row0,
                                            int k0, int ld, int lane) {
  int hi = lane >> 4;
  const _Float16* p = base + (row0 + (lane & 15)) * ld + k0 + hi * 8;
  union { v16h v; v8h h[2]; } u;
  u.h[0] = *(const v8h*)(p);        // K chunk k0..k0+7 (+8 for hi lanes)
  u.h[1] = *(const v8h*)(p + 16);   // K chunk k0+16..k0+23 (+8 for hi lanes)
  return u.v;
}

__device__ __forceinline__ v16h load_b_frag(const _Float16* frag, int lane) {
  return *(const v16h*)(frag + lane * 16);   // 32 contiguous bytes per lane
}

// Fused epilogue. MODE: 0 store f16, 1 relu->f16, 2 h+=relu(v),
// 3 h=relu(h+relu(v)), 4 store f32.
template<int MODE>
__device__ __forceinline__ void epilogue(v8f c, const float* bias, int rt,
                                         int ct, int lane, _Float16* O16,
                                         float* O32, int ldO) {
  int n = ct * 16 + (lane & 15);
  float bs = bias[n];
  int rbase = rt * 16 + ((lane >> 4) << 3);
#pragma unroll
  for (int r = 0; r < 8; ++r) {
    float v = c[r] + bs;
    int row = rbase + r;
    if (MODE == 0) {
      O16[row * ldO + n] = (_Float16)v;
    } else if (MODE == 1) {
      O16[row * ldO + n] = (_Float16)fmaxf(v, 0.f);
    } else if (MODE == 2) {
      float hv = (float)O16[row * ldO + n];
      O16[row * ldO + n] = (_Float16)(hv + fmaxf(v, 0.f));
    } else if (MODE == 3) {
      float hv = (float)O16[row * ldO + n];
      O16[row * ldO + n] = (_Float16)fmaxf(hv + fmaxf(v, 0.f), 0.f);
    } else {
      O32[row * ldO + n] = v;
    }
  }
}

// K=256 GEMM row: hoist all 8 A fragments to registers (64 VGPRs), then
// stream B fragments over `nct` column tiles (8 WMMAs per tile).
template<int MODE>
__device__ __forceinline__ void gemm_row8k(const _Float16* A,
                                           const _Float16* Wl, const float* bias,
                                           int rt, int ct0, int nct, int lane,
                                           _Float16* O16, float* O32, int ldO) {
  v16h afr[8];
#pragma unroll
  for (int kt = 0; kt < 8; ++kt)
    afr[kt] = load_a_frag(A, rt * 16, kt * 32, 256, lane);
#pragma unroll 1
  for (int j = 0; j < nct; ++j) {
    const _Float16* fb = Wl + (size_t)((ct0 + j) * 8) * 512;
    __builtin_prefetch(fb + 8 * 512 + lane * 16, 0, 1);  // next column tile
    v8f c = {};
#pragma unroll
    for (int kt = 0; kt < 8; ++kt) {
      v16h b = load_b_frag(fb + kt * 512, lane);
      c = __builtin_amdgcn_wmma_f32_16x16x32_f16(false, afr[kt], false, b,
                                                 (short)0, c, false, false);
    }
    epilogue<MODE>(c, bias, rt, ct0 + j, lane, O16, O32, ldO);
  }
}

// K=32 input GEMM: one A fragment reused across all 8 column tiles.
__device__ __forceinline__ void gemm0(const _Float16* mxb, const _Float16* WinL,
                                      const float* bias, int rt, int ct0,
                                      int lane, _Float16* O16) {
  v16h a = load_a_frag(mxb, rt * 16, 0, 32, lane);
#pragma unroll 1
  for (int j = 0; j < 8; ++j) {
    v16h b = load_b_frag(WinL + (size_t)(ct0 + j) * 512, lane);
    v8f c = {};
    c = __builtin_amdgcn_wmma_f32_16x16x32_f16(false, a, false, b,
                                               (short)0, c, false, false);
    epilogue<0>(c, bias, rt, ct0 + j, lane, O16, nullptr, 256);
  }
}

__global__ __launch_bounds__(256, 1)
void flow_kernel(const float* __restrict__ x,
                 const float* __restrict__ sc_scale,
                 const float* __restrict__ sc_shift,
                 const float* __restrict__ b_in,  const float* __restrict__ b1a,
                 const float* __restrict__ b1b,   const float* __restrict__ b2a,
                 const float* __restrict__ b2b,   const float* __restrict__ b_out,
                 const int*   __restrict__ perms,
                 const _Float16* __restrict__ Win16,
                 const _Float16* __restrict__ W1a16,
                 const _Float16* __restrict__ W1b16,
                 const _Float16* __restrict__ W2a16,
                 const _Float16* __restrict__ W2b16,
                 const _Float16* __restrict__ Wout16,
                 float* __restrict__ out) {
  extern __shared__ char smem[];
  float*    xb  = (float*)(smem + XB_OFF);
  _Float16* mxb = (_Float16*)(smem + MX_OFF);
  _Float16* hb  = (_Float16*)(smem + HB_OFF);
  _Float16* rb  = (_Float16*)(smem + RB_OFF);
  float*    pb  = (float*)(smem + PB_OFF);
  float*    acc = (float*)(smem + AC_OFF);

  const int tid  = threadIdx.x;
  const int lane = tid & 31;
  const int wave = tid >> 5;
  const int rt   = wave & 3;    // row tile 0..3
  const int cg   = wave >> 2;   // column group 0/1
  const int s0   = blockIdx.x * TB;

  for (int q = tid; q < TB * N_DIM; q += 256)
    xb[q] = x[(long)s0 * N_DIM + q];
  if (tid < TB) acc[tid] = 0.f;
  __syncthreads();

  for (int l = 0; l < L_LAY; ++l) {
    const _Float16* WinL  = Win16  + (size_t)l * 16 * 1 * 512;
    const _Float16* W1aL  = W1a16  + (size_t)l * 16 * 8 * 512;
    const _Float16* W1bL  = W1b16  + (size_t)l * 16 * 8 * 512;
    const _Float16* W2aL  = W2a16  + (size_t)l * 16 * 8 * 512;
    const _Float16* W2bL  = W2b16  + (size_t)l * 16 * 8 * 512;
    const _Float16* WoutL = Wout16 + (size_t)l * 50 * 8 * 512;

    // ---- permute + affine + mask -> mx ----
    float vloc[8];
#pragma unroll
    for (int j = 0; j < 8; ++j) {
      int q = tid + 256 * j;
      int s = q >> 5, n = q & 31;
      vloc[j] = xb[s * N_DIM + perms[l * N_DIM + n]];
    }
    __syncthreads();
#pragma unroll
    for (int j = 0; j < 8; ++j) {
      int q = tid + 256 * j;
      int s = q >> 5, n = q & 31;
      float v = vloc[j] * sc_scale[l * N_DIM + n] + sc_shift[l * N_DIM + n];
      xb[s * N_DIM + n] = v;
      bool kept = (l & 1) ? ((n & 1) == 0) : ((n & 1) == 1);
      mxb[s * N_DIM + n] = (_Float16)(kept ? v : 0.f);
    }
    __syncthreads();

    // ---- h = mx @ W_in + b_in ----
    gemm0(mxb, WinL, b_in + l * H_DIM, rt, cg * 8, lane, hb);
    __syncthreads();

    // ---- residual block 1: h += relu(relu(h@W1a+b1a)@W1b+b1b) ----
    gemm_row8k<1>(hb, W1aL, b1a + l * H_DIM, rt, cg * 8, 8, lane,
                  rb, nullptr, 256);
    __syncthreads();
    gemm_row8k<2>(rb, W1bL, b1b + l * H_DIM, rt, cg * 8, 8, lane,
                  hb, nullptr, 256);
    __syncthreads();

    // ---- residual block 2 (final outer relu folded into MODE 3) ----
    gemm_row8k<1>(hb, W2aL, b2a + l * H_DIM, rt, cg * 8, 8, lane,
                  rb, nullptr, 256);
    __syncthreads();
    gemm_row8k<3>(rb, W2bL, b2b + l * H_DIM, rt, cg * 8, 8, lane,
                  hb, nullptr, 256);
    __syncthreads();

    // ---- params = h @ W_out + b_out (f32 to LDS), 25 col tiles/wave ----
    gemm_row8k<4>(hb, WoutL, b_out + l * NP_, rt, cg * 25, 25, lane,
                  nullptr, pb, NP_);
    __syncthreads();

    // ---- RQ-spline + logdet ----
#pragma unroll 1
    for (int j = 0; j < 8; ++j) {
      int q = tid + 256 * j;
      int s = q >> 5, n = q & 31;
      float contrib = logf(fabsf(sc_scale[l * N_DIM + n]));
      bool kept = (l & 1) ? ((n & 1) == 0) : ((n & 1) == 1);
      if (!kept) {
        const float* pp = pb + s * NP_ + n * P_PAR;
        float xv = xb[s * N_DIM + n];
        float uw[8], uh[8];
#pragma unroll
        for (int k = 0; k < 8; ++k) { uw[k] = pp[k]; uh[k] = pp[8 + k]; }
        float mw = uw[0], mh = uh[0];
#pragma unroll
        for (int k = 1; k < 8; ++k) { mw = fmaxf(mw, uw[k]); mh = fmaxf(mh, uh[k]); }
        float sw = 0.f, sh = 0.f;
#pragma unroll
        for (int k = 0; k < 8; ++k) {
          uw[k] = expf(uw[k] - mw); sw += uw[k];
          uh[k] = expf(uh[k] - mh); sh += uh[k];
        }
        const float span = 20.0f - 8.0f * 1e-4f;
        float xp[9], yp[9];
        xp[0] = -10.f; yp[0] = -10.f;
        float cw = 0.f, ch = 0.f;
#pragma unroll
        for (int k = 0; k < 8; ++k) {
          cw += uw[k] / sw * span + 1e-4f;
          ch += uh[k] / sh * span + 1e-4f;
          xp[k + 1] = -10.f + cw;
          yp[k + 1] = -10.f + ch;
        }
        float dd[9];
#pragma unroll
        for (int k = 0; k < 9; ++k) {
          float zv = pp[16 + k] + 0.54116666f;     // log(expm1(1 - 1e-4))
          dd[k] = log1pf(expf(zv)) + 1e-4f;        // softplus + MIN_SLOPE
        }
        bool inside = (xv > -10.f) && (xv < 10.f);
        float xc = fminf(fmaxf(xv, -10.f), 10.f);
        int idx = 0;
#pragma unroll
        for (int k = 1; k < 9; ++k) idx += (xc >= xp[k]) ? 1 : 0;
        idx = idx > 7 ? 7 : idx;
        float xk = 0, xk1 = 0, yk = 0, yk1 = 0, d0 = 0, d1 = 0;
#pragma unroll
        for (int k = 0; k < 8; ++k)
          if (idx == k) { xk = xp[k]; xk1 = xp[k + 1]; yk = yp[k];
                          yk1 = yp[k + 1]; d0 = dd[k]; d1 = dd[k + 1]; }
        float bw = xk1 - xk, bh = yk1 - yk;
        float sl = bh / bw;
        float z = fminf(fmaxf((xc - xk) / bw, 0.f), 1.f);
        float zz = z * (1.f - z);
        float den = sl + (d1 + d0 - 2.f * sl) * zz;
        float y = yk + bh * (sl * z * z + d0 * zz) / den;
        float ld = 2.f * logf(sl)
                 + logf(d1 * z * z + 2.f * sl * zz + d0 * (1.f - z) * (1.f - z))
                 - 2.f * logf(den);
        if (inside) { xb[s * N_DIM + n] = y; contrib += ld; }
      }
      atomicAdd(&acc[s], contrib);
    }
    __syncthreads();
  }

  // ---- Gaussian log-prob + output ----
#pragma unroll
  for (int j = 0; j < 8; ++j) {
    int q = tid + 256 * j;
    float xv = xb[q];
    atomicAdd(&acc[q >> 5], -0.5f * xv * xv);
  }
  __syncthreads();
  if (tid < TB) out[s0 + tid] = acc[tid] - 29.406033f;  // 0.5*N*log(2*pi)
}

extern "C" void kernel_launch(void* const* d_in, const int* in_sizes, int n_in,
                              void* d_out, int out_size, void* d_ws, size_t ws_size,
                              hipStream_t stream) {
  const float* x        = (const float*)d_in[0];
  const float* sc_scale = (const float*)d_in[1];
  const float* sc_shift = (const float*)d_in[2];
  const float* W_in     = (const float*)d_in[3];
  const float* b_in     = (const float*)d_in[4];
  const float* W1a      = (const float*)d_in[5];
  const float* b1a      = (const float*)d_in[6];
  const float* W1b      = (const float*)d_in[7];
  const float* b1b      = (const float*)d_in[8];
  const float* W2a      = (const float*)d_in[9];
  const float* b2a      = (const float*)d_in[10];
  const float* W2b      = (const float*)d_in[11];
  const float* b2b      = (const float*)d_in[12];
  const float* W_out    = (const float*)d_in[13];
  const float* b_out    = (const float*)d_in[14];
  const int*   perms    = (const int*)d_in[15];
  float* out = (float*)d_out;

  // carve f16 weight fragments out of workspace (~7.3 MB)
  _Float16* w = (_Float16*)d_ws;
  size_t o = 0;
  _Float16* Win16  = w + o; o += (size_t)L_LAY * 32 * 256;
  _Float16* W1a16  = w + o; o += (size_t)L_LAY * 256 * 256;
  _Float16* W1b16  = w + o; o += (size_t)L_LAY * 256 * 256;
  _Float16* W2a16  = w + o; o += (size_t)L_LAY * 256 * 256;
  _Float16* W2b16  = w + o; o += (size_t)L_LAY * 256 * 256;
  _Float16* Wout16 = w + o;

  struct Conv { const float* src; _Float16* dst; int hin, hout; } convs[6] = {
    { W_in,  Win16,  32,  256 }, { W1a, W1a16, 256, 256 },
    { W1b, W1b16, 256, 256 },    { W2a, W2a16, 256, 256 },
    { W2b, W2b16, 256, 256 },    { W_out, Wout16, 256, 800 },
  };
  for (int i = 0; i < 6; ++i) {
    long total = (long)L_LAY * convs[i].hin * convs[i].hout;
    int blocks = (int)((total + 255) / 256);
    wconv<<<blocks, 256, 0, stream>>>(convs[i].src, convs[i].dst,
                                      convs[i].hin, convs[i].hout);
  }

  flow_kernel<<<B_TOT / TB, 256, SMEM_BYTES, stream>>>(
      x, sc_scale, sc_shift, b_in, b1a, b1b, b2a, b2b, b_out, perms,
      Win16, W1a16, W1b16, W2a16, W2b16, Wout16, out);
}
